// InvariantParticleAttention_75771813036631
// MI455X (gfx1250) — matrix operand order repaired and checked by reading, not compiled, not run on hardware
//
#include <hip/hip_runtime.h>

// Problem constants (fixed by the reference harness)
#define B_DIM 8
#define H_DIM 8
#define N_DIM 1024
#define C_DIM 128
#define BM    128   // query rows per workgroup
#define BK    64    // key rows per inner block
#define NTHREADS 256
#define NBLK  (N_DIM / BK)

// LDS strides (halves), padded to 16B multiples to keep v8h alignment and
// rotate banks (stride 272B = 68 banks -> 4-bank shift per row).
#define QS_STR 136
#define KS_STR 136
#define VT_STR 72
#define PS_STR 72
#define OS_STR 132  // floats

typedef _Float16 v16h __attribute__((ext_vector_type(16)));
typedef _Float16 v8h  __attribute__((ext_vector_type(8)));
typedef float    v8f  __attribute__((ext_vector_type(8)));

union AF { v16h v; v8h h[2]; };

// inv[i][l]      = eta_i * eta_l * L[l][i]      (eta = diag(1,-1,-1,-1))
__device__ __forceinline__ void mat_inv(const float* __restrict__ L16, float M[4][4]) {
  const float eta[4] = {1.f, -1.f, -1.f, -1.f};
  #pragma unroll
  for (int i = 0; i < 4; ++i)
    #pragma unroll
    for (int l = 0; l < 4; ++l)
      M[i][l] = eta[i] * eta[l] * L16[l * 4 + i];
}
// lower_inv[i][k] = eta_k * L[k][i]
__device__ __forceinline__ void mat_low(const float* __restrict__ L16, float M[4][4]) {
  const float eta[4] = {1.f, -1.f, -1.f, -1.f};
  #pragma unroll
  for (int i = 0; i < 4; ++i)
    #pragma unroll
    for (int kk = 0; kk < 4; ++kk)
      M[i][kk] = eta[kk] * L16[kk * 4 + i];
}
// forward: M = L
__device__ __forceinline__ void mat_fwd(const float* __restrict__ L16, float M[4][4]) {
  #pragma unroll
  for (int i = 0; i < 4; ++i)
    #pragma unroll
    for (int j = 0; j < 4; ++j)
      M[i][j] = L16[i * 4 + j];
}
__device__ __forceinline__ void xf4(const float M[4][4], float y[4]) {
  float x0 = y[0], x1 = y[1], x2 = y[2], x3 = y[3];
  #pragma unroll
  for (int i = 0; i < 4; ++i)
    y[i] = M[i][0] * x0 + M[i][1] * x1 + M[i][2] * x2 + M[i][3] * x3;
}

__global__ __launch_bounds__(NTHREADS)
void ipa_flash_wmma_kernel(const float* __restrict__ q,
                           const float* __restrict__ k,
                           const float* __restrict__ v,
                           const float* __restrict__ L,
                           float* __restrict__ out) {
  // 89,088 bytes of LDS; epilogue reuses it as 128x132 f32 (67,584 B).
  __shared__ __align__(16) _Float16 smem[44544];
  _Float16* Qs = smem;           // 128 x 136 halves
  _Float16* Ks = smem + 17408;   //  64 x 136
  _Float16* Vt = smem + 26112;   // 128 x 72   (transposed: [c][key])
  _Float16* Ps = smem + 35328;   // 8 waves x 16 x 72
  float*    Os = reinterpret_cast<float*>(smem);

  const int tid  = threadIdx.x;
  const int lane = tid & 31;
  const int wave = tid >> 5;
  const int l15  = lane & 15;
  const int hi   = lane >> 4;           // 0 or 1: which 16-lane half

  const int bh    = blockIdx.x >> 3;    // 0..63  (b*H + h)
  const int qblk  = blockIdx.x & 7;     // 0..7
  const int b     = bh >> 3;            // H == 8
  const int qbase = qblk * BM;

  const size_t base = (size_t)bh * N_DIM * C_DIM;
  const float  invs = 0.08838834764831843f;  // 1/sqrt(128), folded into Q

  // ---------------- Q: transform by inv(L), scale, f16 -> LDS -------------
  for (int task = tid; task < BM * 32; task += NTHREADS) {
    const int m = task >> 5, j = task & 31;   // j = 4-channel group
    const int n = qbase + m;
    float4 x = *reinterpret_cast<const float4*>(q + base + (size_t)n * C_DIM + 4 * j);
    float y[4] = {x.x, x.y, x.z, x.w};
    if (j >= 2) {  // groups 0,1 are the 8 scalar channels (identity)
      const float* L16 = L + (size_t)(b * N_DIM + n) * 16;
      float M[4][4]; mat_inv(L16, M); xf4(M, y);
    }
    _Float16* d = &Qs[m * QS_STR + 4 * j];
    d[0] = (_Float16)(y[0] * invs); d[1] = (_Float16)(y[1] * invs);
    d[2] = (_Float16)(y[2] * invs); d[3] = (_Float16)(y[3] * invs);
  }
  __syncthreads();

  // ---------------- Q A-fragments (16x32 f16 layout) ----------------------
  v16h qf[4];
  {
    const _Float16* qr = &Qs[(wave * 16 + l15) * QS_STR];
    #pragma unroll
    for (int kc = 0; kc < 4; ++kc) {
      AF a;
      const int c0 = 32 * kc + (hi ? 16 : 0);
      a.h[0] = *reinterpret_cast<const v8h*>(qr + c0);        // K = c0..c0+7
      a.h[1] = *reinterpret_cast<const v8h*>(qr + c0 + 8);    // K = c0+8..
      qf[kc] = a.v;
    }
  }

  float rmax[8], rsum[8];
  v8f o[8];
  const v8f vzero = {};
  #pragma unroll
  for (int r = 0; r < 8; ++r) { rmax[r] = -3.0e38f; rsum[r] = 0.f; }
  #pragma unroll
  for (int cc = 0; cc < 8; ++cc) o[cc] = vzero;

  // ================= main loop over key blocks ============================
  for (int jb = 0; jb < NBLK; ++jb) {
    __syncthreads();   // previous block's K/V reads done before overwrite
    const int kbase = jb * BK;

    // K: lower_inv transform, row-major; V: inv transform, transposed.
    for (int task = tid; task < BK * 32; task += NTHREADS) {
      const int r = task >> 5, j = task & 31;
      const int n = kbase + r;
      float4 xk = *reinterpret_cast<const float4*>(k + base + (size_t)n * C_DIM + 4 * j);
      float4 xv = *reinterpret_cast<const float4*>(v + base + (size_t)n * C_DIM + 4 * j);
      float yk[4] = {xk.x, xk.y, xk.z, xk.w};
      float yv[4] = {xv.x, xv.y, xv.z, xv.w};
      if (j >= 2) {
        const float* L16 = L + (size_t)(b * N_DIM + n) * 16;
        float Ml[4][4], Mi[4][4];
        mat_low(L16, Ml); mat_inv(L16, Mi);
        xf4(Ml, yk); xf4(Mi, yv);
      }
      _Float16* dk = &Ks[r * KS_STR + 4 * j];
      dk[0] = (_Float16)yk[0]; dk[1] = (_Float16)yk[1];
      dk[2] = (_Float16)yk[2]; dk[3] = (_Float16)yk[3];
      #pragma unroll
      for (int c = 0; c < 4; ++c)
        Vt[(4 * j + c) * VT_STR + r] = (_Float16)yv[c];
    }
    __syncthreads();

    // L2 warm-up for next key block while this block computes
    // (lowers to global_prefetch_b8; one 128B line per thread per tensor).
    if (jb + 1 < NBLK) {
      const size_t poff = base + (size_t)(kbase + BK) * C_DIM + (size_t)tid * 32;
      __builtin_prefetch(k + poff, 0, 1);
      __builtin_prefetch(v + poff, 0, 1);
    }

    // ---------------- S = Q * K^T  (4 tiles, chunk-outer, 2-deep pipeline)
    v8f s[4];
    #pragma unroll
    for (int t = 0; t < 4; ++t) s[t] = vzero;
    {
      const _Float16* krow[4];
      #pragma unroll
      for (int t = 0; t < 4; ++t) krow[t] = &Ks[(t * 16 + l15) * KS_STR];
      const int cb0 = hi ? 16 : 0;
      AF cb[2][4];
      #pragma unroll
      for (int t = 0; t < 4; ++t) {            // preload chunk 0
        cb[0][t].h[0] = *reinterpret_cast<const v8h*>(krow[t] + cb0);
        cb[0][t].h[1] = *reinterpret_cast<const v8h*>(krow[t] + cb0 + 8);
      }
      #pragma unroll
      for (int kc = 0; kc < 4; ++kc) {
        if (kc < 3) {                          // prefetch chunk kc+1
          const int c1 = 32 * (kc + 1) + cb0;
          #pragma unroll
          for (int t = 0; t < 4; ++t) {
            cb[(kc + 1) & 1][t].h[0] = *reinterpret_cast<const v8h*>(krow[t] + c1);
            cb[(kc + 1) & 1][t].h[1] = *reinterpret_cast<const v8h*>(krow[t] + c1 + 8);
          }
        }
        #pragma unroll
        for (int t = 0; t < 4; ++t)            // 4 independent-D WMMAs
          s[t] = __builtin_amdgcn_wmma_f32_16x16x32_f16(
              false, qf[kc], false, cb[kc & 1][t].v, (short)0, s[t], false, false);
      }
    }

    // ---------------- online softmax (rows live in one 16-lane half) ------
    float al[8];
    #pragma unroll
    for (int r = 0; r < 8; ++r) {
      float m0 = fmaxf(fmaxf(s[0][r], s[1][r]), fmaxf(s[2][r], s[3][r]));
      #pragma unroll
      for (int off = 1; off < 16; off <<= 1)
        m0 = fmaxf(m0, __shfl_xor(m0, off, 32));
      const float nm = fmaxf(rmax[r], m0);
      al[r] = __expf(rmax[r] - nm);
      rmax[r] = nm;
    }
    #pragma unroll
    for (int r = 0; r < 8; ++r) {
      float ps = 0.f;
      #pragma unroll
      for (int t = 0; t < 4; ++t) {
        const float p = __expf(s[t][r] - rmax[r]);
        s[t][r] = p; ps += p;
      }
      #pragma unroll
      for (int off = 1; off < 16; off <<= 1)
        ps += __shfl_xor(ps, off, 32);
      rsum[r] = rsum[r] * al[r] + ps;
    }
    #pragma unroll
    for (int cc = 0; cc < 8; ++cc)
      #pragma unroll
      for (int r = 0; r < 8; ++r)
        o[cc][r] = o[cc][r] * al[r];

    // ---------------- P -> LDS (re-swizzle C/D layout -> A layout) --------
    {
      _Float16* pw = &Ps[wave * 16 * PS_STR];
      #pragma unroll
      for (int t = 0; t < 4; ++t)
        #pragma unroll
        for (int r = 0; r < 8; ++r)
          pw[(r + 8 * hi) * PS_STR + t * 16 + l15] = (_Float16)s[t][r];
    }
    v16h pf[2];
    {
      const _Float16* pr = &Ps[(wave * 16 + l15) * PS_STR];
      #pragma unroll
      for (int kk = 0; kk < 2; ++kk) {
        AF a;
        const int c0 = 32 * kk + (hi ? 16 : 0);
        a.h[0] = *reinterpret_cast<const v8h*>(pr + c0);
        a.h[1] = *reinterpret_cast<const v8h*>(pr + c0 + 8);
        pf[kk] = a.v;
      }
    }

    // ---------------- O += P * V  (16 steps, 1-step-ahead pipeline) -------
    {
      const int cv = hi ? 16 : 0;
      AF vb[2];
      vb[0].h[0] = *reinterpret_cast<const v8h*>(&Vt[l15 * VT_STR] + cv);
      vb[0].h[1] = *reinterpret_cast<const v8h*>(&Vt[l15 * VT_STR] + cv + 8);
      #pragma unroll
      for (int idx = 0; idx < 16; ++idx) {     // idx = cc*2 + kk
        const int cc = idx >> 1, kk = idx & 1;
        if (idx < 15) {
          const int ni = idx + 1;
          const _Float16* vr = &Vt[((ni >> 1) * 16 + l15) * VT_STR];
          const int c0 = 32 * (ni & 1) + cv;
          vb[ni & 1].h[0] = *reinterpret_cast<const v8h*>(vr + c0);
          vb[ni & 1].h[1] = *reinterpret_cast<const v8h*>(vr + c0 + 8);
        }
        o[cc] = __builtin_amdgcn_wmma_f32_16x16x32_f16(
            false, pf[kk], false, vb[idx & 1].v, (short)0, o[cc], false, false);
      }
    }
  }

  // ---------------- normalize, stage O in LDS (reuse smem) ----------------
  __syncthreads();
  #pragma unroll
  for (int r = 0; r < 8; ++r) {
    const float inv_l = 1.0f / rsum[r];
    const int m = wave * 16 + r + 8 * hi;
    #pragma unroll
    for (int cc = 0; cc < 8; ++cc)
      Os[m * OS_STR + cc * 16 + l15] = o[cc][r] * inv_l;
  }
  __syncthreads();

  // ---------------- output transform by forward L, fp32 store -------------
  for (int task = tid; task < BM * 32; task += NTHREADS) {
    const int m = task >> 5, j = task & 31;
    const int n = qbase + m;
    float4 x = *reinterpret_cast<const float4*>(&Os[m * OS_STR + 4 * j]);
    float y[4] = {x.x, x.y, x.z, x.w};
    if (j >= 2) {
      const float* L16 = L + (size_t)(b * N_DIM + n) * 16;
      float M[4][4]; mat_fwd(L16, M); xf4(M, y);
    }
    float4 r4 = make_float4(y[0], y[1], y[2], y[3]);
    *reinterpret_cast<float4*>(out + base + (size_t)n * C_DIM + 4 * j) = r4;
  }
}

extern "C" void kernel_launch(void* const* d_in, const int* in_sizes, int n_in,
                              void* d_out, int out_size, void* d_ws, size_t ws_size,
                              hipStream_t stream) {
  (void)in_sizes; (void)n_in; (void)out_size; (void)d_ws; (void)ws_size;
  const float* q = (const float*)d_in[0];
  const float* k = (const float*)d_in[1];
  const float* v = (const float*)d_in[2];
  const float* L = (const float*)d_in[3];
  float* out = (float*)d_out;

  const int grid = B_DIM * H_DIM * (N_DIM / BM);  // 512 workgroups
  ipa_flash_wmma_kernel<<<dim3(grid), dim3(NTHREADS), 0, stream>>>(q, k, v, L, out);
}
